// OpenAIMoeSparseMoeBlock_34136400068634
// MI455X (gfx1250) — compile-verified, matrix operand order checked by reading, not run on hardware
//
#include <hip/hip_runtime.h>
#include <hip/hip_bf16.h>

// Problem constants (from reference): E=32 experts, K=4 topk, H=I=T=1024
#define E_ 32
#define K_ 4
#define H_ 1024
#define I_ 1024
#define T_ 1024
#define TWO_I 2048
#define ALPHA_ 1.702f
#define BETA_ 1.0f
#define LDA_ 1032   // padded bf16 row stride: 2064B = 129*16B -> 16B aligned,
                    // dword-stride 516 === 4 (mod 64) -> 16 rows hit 16 distinct bank groups

typedef __attribute__((ext_vector_type(16))) __bf16 v16bf;
typedef __attribute__((ext_vector_type(8)))  float  v8f;

union ABfrag { v16bf v; uint4 q[2]; __bf16 h[16]; };

// ---------------------------------------------------------------------------
// Kernel 0: zero output accumulator and per-expert dispatch counters
// ---------------------------------------------------------------------------
__global__ void moe_zero_kernel(float* __restrict__ out, int* __restrict__ counts) {
    int i = blockIdx.x * blockDim.x + threadIdx.x;
    if (i < T_ * H_) out[i] = 0.0f;
    if (i < E_)      counts[i] = 0;
}

// ---------------------------------------------------------------------------
// Kernel 1: router — one wave32 per token, lane == expert.
// softmax + top-4 + renorm entirely with wave32 shuffles; lane0 builds
// per-expert dispatch lists (token index + combine weight).
// ---------------------------------------------------------------------------
__global__ __launch_bounds__(32) void moe_router_kernel(
    const float* __restrict__ hs, const float* __restrict__ gw,
    const float* __restrict__ gb, int* __restrict__ counts,
    int* __restrict__ tok_list, float* __restrict__ w_list)
{
    const int t    = blockIdx.x;
    const int lane = threadIdx.x;           // lane == expert id (E_ == 32 == wave32)

    const float4* hs4 = (const float4*)(hs + (size_t)t * H_);
    float acc = 0.0f;
    for (int h4 = 0; h4 < H_ / 4; ++h4) {
        float4 v = hs4[h4];                  // broadcast across lanes
        int h = h4 * 4;
        acc += v.x * gw[(h + 0) * E_ + lane];   // coalesced across lanes
        acc += v.y * gw[(h + 1) * E_ + lane];
        acc += v.z * gw[(h + 2) * E_ + lane];
        acc += v.w * gw[(h + 3) * E_ + lane];
    }
    float logit = acc + gb[lane];

    // wave-wide softmax
    float m = logit;
    for (int off = 16; off > 0; off >>= 1) m = fmaxf(m, __shfl_xor(m, off));
    float p = __expf(logit - m);
    float s = p;
    for (int off = 16; off > 0; off >>= 1) s += __shfl_xor(s, off);
    p /= s;

    // iterative top-4 argmax (ties -> lowest index, matches top_k behavior)
    float pv = p;
    int   selid[K_];
    float selv[K_];
    float wsum = 0.0f;
    for (int k = 0; k < K_; ++k) {
        float v = pv; int id = lane;
        for (int off = 16; off > 0; off >>= 1) {
            float ov = __shfl_xor(v, off);
            int  oid = __shfl_xor(id, off);
            if (ov > v || (ov == v && oid < id)) { v = ov; id = oid; }
        }
        selid[k] = id; selv[k] = v; wsum += v;
        if (lane == id) pv = -1.0f;
    }

    if (lane == 0) {
        for (int k = 0; k < K_; ++k) {
            int   e = selid[k];
            float w = selv[k] / wsum;
            int slot = atomicAdd(counts + e, 1);
            tok_list[e * T_ + slot] = t;
            w_list [e * T_ + slot] = w;
        }
    }
}

// ---------------------------------------------------------------------------
// Kernel 2: expert MLP. Block = 256 threads = 8 waves, tile = 16 tokens.
// Phase A: h = tok(16xH) @ w1[e](Hx2I) via v_wmma_f32_16x16x32_bf16,
//          bias + (up+B)*gate*sigmoid(A*gate) fused via shfl pairing -> LDS.
// Phase B: y = act(16xI) @ w2[e](IxH), +b2, *combine_w, atomicAdd into out.
// Weights converted fp32->bf16 in flight (memory-bound; cvt hides under HBM).
// ---------------------------------------------------------------------------
__global__ __launch_bounds__(256) void moe_expert_kernel(
    const float* __restrict__ hs, const float* __restrict__ w1,
    const float* __restrict__ b1, const float* __restrict__ w2,
    const float* __restrict__ b2, const int* __restrict__ counts,
    const int* __restrict__ tok_list, const float* __restrict__ w_list,
    float* __restrict__ out)
{
    __shared__ __bf16 tokA[16][LDA_];   // 16 token rows, bf16
    __shared__ __bf16 actS[16][LDA_];   // 16 activation rows, bf16
    __shared__ int    tokS[16];
    __shared__ float  cwS[16];

    const int e       = blockIdx.y;
    const int tile    = blockIdx.x;
    const int ne      = counts[e];
    const int rowbase = tile * 16;
    if (rowbase >= ne) return;          // empty tile for this expert

    const int tid  = threadIdx.x;
    const int lane = tid & 31;
    const int wave = tid >> 5;          // 0..7
    const int col  = lane & 15;         // N column (B/C/D) and M row (A) for this lane
    const int half = lane >> 4;         // 0 or 1
    const int koff = half * 8;          // ISA 16-bit A/B fragment K offset per lane half

    const float* w1e = w1 + (size_t)e * H_ * TWO_I;
    const float* w2e = w2 + (size_t)e * I_ * H_;

    // --- stage dispatch metadata ---
    if (tid < 16) {
        int   m  = rowbase + tid;
        tokS[tid] = (m < ne) ? tok_list[e * T_ + m] : 0;
        cwS[tid]  = (m < ne) ? w_list [e * T_ + m] : 0.0f;  // pad rows contribute 0
    }
    __syncthreads();

    // --- gather 16 token rows fp32 -> bf16 into LDS (float4 global loads) ---
    for (int idx = tid; idx < 16 * (H_ / 4); idx += 256) {
        int row = idx >> 8;             // 256 float4 chunks per row
        int h   = (idx & 255) * 4;
        float4 v = *(const float4*)(hs + (size_t)tokS[row] * H_ + h);
        __bf16* dst = &tokA[row][h];
        dst[0] = (__bf16)v.x; dst[1] = (__bf16)v.y;
        dst[2] = (__bf16)v.z; dst[3] = (__bf16)v.w;
    }
    __syncthreads();

    // =========================== Phase A ===================================
    // 2I/16 = 128 N-tiles across 8 waves -> 16 tiles each
    for (int i = 0; i < 16; ++i) {
        const int tn   = wave + 8 * i;
        const int d0   = tn * 16;
        const int dcol = d0 + col;
        v8f acc = {};
        for (int kb = 0; kb < 32; ++kb) {          // K = 32*32 = 1024
            const int kbase = kb * 32;
            ABfrag a, b;
            a.q[0] = *(const uint4*)&tokA[col][kbase + koff];
            a.q[1] = *(const uint4*)&tokA[col][kbase + 16 + koff];
            const float* wp = w1e + (size_t)(kbase + koff) * TWO_I + dcol;
            #pragma unroll
            for (int j = 0; j < 8; ++j) {
                b.h[j]     = (__bf16)wp[(size_t)j * TWO_I];
                b.h[8 + j] = (__bf16)wp[(size_t)(16 + j) * TWO_I];
            }
            acc = __builtin_amdgcn_wmma_f32_16x16x32_bf16(
                      false, a.v, false, b.v, (short)0, acc, false, false);
        }
        // bias + gated activation; gate/up are even/odd N -> adjacent lanes
        const float bias = b1[e * TWO_I + dcol];
        const int   icol = (d0 >> 1) + (col >> 1);
        #pragma unroll
        for (int r = 0; r < 8; ++r) {
            float x = acc[r] + bias;
            float o = __shfl_xor(x, 1);            // partner column value
            if ((lane & 1) == 0) {                 // even lane holds gate column
                float g   = x, u = o;
                float act = (u + BETA_) * g * (1.0f / (1.0f + __expf(-ALPHA_ * g)));
                actS[r + 8 * half][icol] = (__bf16)act;
            }
        }
    }
    __syncthreads();

    // =========================== Phase B ===================================
    int   tk[8];
    float cw[8];
    #pragma unroll
    for (int r = 0; r < 8; ++r) {
        tk[r] = tokS[r + 8 * half];
        cw[r] = cwS[r + 8 * half];
    }

    // H/16 = 64 N-tiles across 8 waves -> 8 tiles each
    for (int i = 0; i < 8; ++i) {
        const int tn   = wave + 8 * i;
        const int d0   = tn * 16;
        const int hcol = d0 + col;
        v8f acc = {};
        for (int kb = 0; kb < 32; ++kb) {          // K = I = 1024
            const int kbase = kb * 32;
            ABfrag a, b;
            a.q[0] = *(const uint4*)&actS[col][kbase + koff];
            a.q[1] = *(const uint4*)&actS[col][kbase + 16 + koff];
            const float* wp = w2e + (size_t)(kbase + koff) * H_ + hcol;
            #pragma unroll
            for (int j = 0; j < 8; ++j) {
                b.h[j]     = (__bf16)wp[(size_t)j * H_];
                b.h[8 + j] = (__bf16)wp[(size_t)(16 + j) * H_];
            }
            acc = __builtin_amdgcn_wmma_f32_16x16x32_bf16(
                      false, a.v, false, b.v, (short)0, acc, false, false);
        }
        const float bias = b2[e * H_ + hcol];
        #pragma unroll
        for (int r = 0; r < 8; ++r) {
            float y = (acc[r] + bias) * cw[r];
            atomicAdd(out + (size_t)tk[r] * H_ + hcol, y);
        }
    }
}

// ---------------------------------------------------------------------------
extern "C" void kernel_launch(void* const* d_in, const int* in_sizes, int n_in,
                              void* d_out, int out_size, void* d_ws, size_t ws_size,
                              hipStream_t stream) {
    (void)in_sizes; (void)n_in; (void)out_size; (void)ws_size;
    const float* hs  = (const float*)d_in[0];   // (T,H)
    const float* gw  = (const float*)d_in[1];   // (H,E)
    const float* gb  = (const float*)d_in[2];   // (E)
    const float* w1  = (const float*)d_in[3];   // (E,H,2I)
    const float* b1  = (const float*)d_in[4];   // (E,2I)
    const float* w2  = (const float*)d_in[5];   // (E,I,H)
    const float* b2  = (const float*)d_in[6];   // (E,H)
    float* out = (float*)d_out;                 // (T,H)

    char*  ws       = (char*)d_ws;
    int*   counts   = (int*)ws;                                   // E ints
    int*   tok_list = (int*)(ws + 128);                           // E*T ints
    float* w_list   = (float*)(ws + 128 + (size_t)E_ * T_ * 4);   // E*T floats

    moe_zero_kernel<<<(T_ * H_ + 255) / 256, 256, 0, stream>>>(out, counts);
    moe_router_kernel<<<T_, 32, 0, stream>>>(hs, gw, gb, counts, tok_list, w_list);
    dim3 grid(T_ / 16, E_);   // tiles fastest -> same-expert blocks co-resident -> L2 reuse
    moe_expert_kernel<<<grid, 256, 0, stream>>>(hs, w1, b1, w2, b2,
                                                counts, tok_list, w_list, out);
}
